// Linear_8864812499634
// MI455X (gfx1250) — matrix operand order repaired and checked by reading, not compiled, not run on hardware
//
#include <hip/hip_runtime.h>
#include <stdint.h>

// ---- fixed problem dims from the reference setup ----
#define T_TOK 8192
#define D_DIM 4096
#define O_DIM 4096
#define E_EXP 8
#define R_RANK 32
#define C_DIM 256          // E * R
#define SCALE_F 2.0f       // lora_alpha / r = 64/32

typedef __attribute__((ext_vector_type(8)))  unsigned short v8u;
typedef __attribute__((ext_vector_type(16))) unsigned short v16u;
typedef __attribute__((ext_vector_type(16))) __bf16         v16bf;
typedef __attribute__((ext_vector_type(8)))  float          v8f;
typedef __attribute__((ext_vector_type(4)))  unsigned int   v4u32;
typedef __attribute__((ext_vector_type(8)))  unsigned int   v8u32;

union BF16Frag { v16u u; v16bf bf; };

__device__ __forceinline__ unsigned short f2bf(float x) {
  unsigned u = __builtin_bit_cast(unsigned, x);
  u += 0x7FFFu + ((u >> 16) & 1u);        // round-to-nearest-even
  return (unsigned short)(u >> 16);
}

// ---------------- conversion / packing kernels ----------------
__global__ void cvt_f32_bf16(const float* __restrict__ src,
                             unsigned short* __restrict__ dst, int n4) {
  int i = blockIdx.x * blockDim.x + threadIdx.x;
  int stride = gridDim.x * blockDim.x;
  for (; i < n4; i += stride) {
    float4 v = ((const float4*)src)[i];
    ushort4 o;
    o.x = f2bf(v.x); o.y = f2bf(v.y); o.z = f2bf(v.z); o.w = f2bf(v.w);
    ((ushort4*)dst)[i] = o;
  }
}

// lora_b [E, O, R] -> lbp [O, E*R] bf16
__global__ void pack_lora_b(const float* __restrict__ lb,
                            unsigned short* __restrict__ lbp) {
  int i = blockIdx.x * blockDim.x + threadIdx.x;
  if (i >= E_EXP * O_DIM * R_RANK) return;
  int e   = i / (O_DIM * R_RANK);
  int rem = i % (O_DIM * R_RANK);
  int o   = rem / R_RANK;
  int r   = rem % R_RANK;
  lbp[(size_t)o * C_DIM + e * R_RANK + r] = f2bf(lb[i]);
}

// ---------------- router: softmax + top-2 + renorm, SCALING folded in ----------------
__global__ void __launch_bounds__(256)
router_gates(const float* __restrict__ x, const float* __restrict__ rw,
             const float* __restrict__ rb, float* __restrict__ gates) {
  const int lane = threadIdx.x & 31;
  const int wave = threadIdx.x >> 5;
  const int t = blockIdx.x * 8 + wave;       // one wave32 per token
  float acc[E_EXP];
#pragma unroll
  for (int e = 0; e < E_EXP; ++e) acc[e] = 0.f;
  const float* xp = x + (size_t)t * D_DIM;
  for (int i = lane; i < D_DIM; i += 32) {   // coalesced across lanes
    float xv = xp[i];
#pragma unroll
    for (int e = 0; e < E_EXP; ++e) acc[e] += xv * rw[e * D_DIM + i];
  }
#pragma unroll
  for (int e = 0; e < E_EXP; ++e) {
#pragma unroll
    for (int off = 16; off > 0; off >>= 1) acc[e] += __shfl_xor(acc[e], off, 32);
    acc[e] += rb[e];
  }
  float m = acc[0];
#pragma unroll
  for (int e = 1; e < E_EXP; ++e) m = fmaxf(m, acc[e]);
  float p[E_EXP];
#pragma unroll
  for (int e = 0; e < E_EXP; ++e) p[e] = __expf(acc[e] - m);
  int i1 = 0;
#pragma unroll
  for (int e = 1; e < E_EXP; ++e) if (p[e] > p[i1]) i1 = e;
  int i2 = (i1 == 0) ? 1 : 0;
#pragma unroll
  for (int e = 0; e < E_EXP; ++e) if (e != i1 && p[e] > p[i2]) i2 = e;
  float inv = SCALE_F / (p[i1] + p[i2]);     // softmax denom cancels in renorm
  if (lane < E_EXP) {
    float g = (lane == i1) ? p[i1] * inv : (lane == i2) ? p[i2] * inv : 0.f;
    gates[(size_t)t * E_EXP + lane] = g;
  }
}

// ---------------- WMMA bf16 GEMM machinery ----------------
#define BM 128
#define BN 128
#define BK 32
#define LDT 40   // LDS row stride (elements): 64B row + 16B TDM pad = 80B

// A frag 16x32 bf16: lane r (0..15), kh = lane>>4; K chunks {8kh..8kh+7, 16+8kh..23+8kh}
__device__ __forceinline__ v16bf frag_a(const unsigned short* s, int row, int kh) {
  v8u lo = *(const v8u*)(s + row * LDT + 8 * kh);
  v8u hi = *(const v8u*)(s + row * LDT + 16 + 8 * kh);
  BF16Frag f;
  f.u = __builtin_shufflevector(lo, hi, 0,1,2,3,4,5,6,7,8,9,10,11,12,13,14,15);
  return f.bf;
}
// B frag 32x16 bf16 (from row-major W[o, k]): lane n, kh = lane>>4; K chunks {8kh..8kh+7, 16+8kh..23+8kh}
__device__ __forceinline__ v16bf frag_b(const unsigned short* s, int row, int kh) {
  v8u lo = *(const v8u*)(s + row * LDT + 8 * kh);
  v8u hi = *(const v8u*)(s + row * LDT + 16 + 8 * kh);
  BF16Frag f;
  f.u = __builtin_shufflevector(lo, hi, 0,1,2,3,4,5,6,7,8,9,10,11,12,13,14,15);
  return f.bf;
}

// one BK step: 4x2 WMMA tiles per wave (wave tile 64x32)
__device__ __forceinline__ void mma_step(const unsigned short* As, const unsigned short* Bs,
                                         int wm, int wn, int r, int kh, v8f acc[4][2]) {
  v16bf a[4], b[2];
#pragma unroll
  for (int mi = 0; mi < 4; ++mi) a[mi] = frag_a(As, wm * 64 + mi * 16 + r, kh);
#pragma unroll
  for (int ni = 0; ni < 2; ++ni) b[ni] = frag_b(Bs, wn * 32 + ni * 16 + r, kh);
#pragma unroll
  for (int mi = 0; mi < 4; ++mi)
#pragma unroll
    for (int ni = 0; ni < 2; ++ni)
      acc[mi][ni] = __builtin_amdgcn_wmma_f32_16x16x32_bf16(
          false, a[mi], false, b[ni], (short)0, acc[mi][ni], false, false);
}

// ---- Tensor Data Mover: DMA one 128x32 bf16 tile (row-major, ld elements) into LDS,
//      inserting 16B of pad after every 64B row -> LDS row stride = 80B (LDT elems).
__device__ __forceinline__ void tdm_tile(const unsigned short* g, int ld, unsigned lds_byte) {
  uint64_t ga = (uint64_t)(uintptr_t)g;
  v4u32 g0;
  g0[0] = 1u;                                            // count=1, user descriptor
  g0[1] = lds_byte;                                      // lds_addr (bytes)
  g0[2] = (unsigned)ga;                                  // global_addr[31:0]
  g0[3] = (unsigned)((ga >> 32) & 0x01FFFFFFu) | (2u << 30);  // addr[56:32] | type=2
  v8u32 g1;
  // data_size=2B (1<<16) | pad_enable (1<<20) | pad_interval=16 DW (3<<22) | pad_amount=4 DW (3<<25)
  g1[0] = (1u << 16) | (1u << 20) | (3u << 22) | (3u << 25);
  g1[1] = ((unsigned)ld & 0xFFFFu) << 16;                // tensor_dim0[15:0]   @bits 63:48
  g1[2] = (((unsigned)ld >> 16) & 0xFFFFu) | (128u << 16); // tensor_dim0[31:16] | tensor_dim1[15:0]=128
  g1[3] = ((unsigned)BK) << 16;                          // tensor_dim1[31:16]=0 | tile_dim0=32
  g1[4] = 128u;                                          // tile_dim1=128 | tile_dim2=0
  g1[5] = (unsigned)ld;                                  // tensor_dim0_stride[31:0]
  g1[6] = 0u;                                            // stride[47:32]=0 | tensor_dim1_stride lo
  g1[7] = 0u;
  asm volatile("tensor_load_to_lds %0, %1" :: "s"(g0), "s"(g1) : "memory");
}

#define TDM_WAIT() __builtin_amdgcn_s_wait_tensorcnt(0)

// h = xb @ la.T  (M=T, N=256, K=4096), epilogue applies gates -> gh bf16
__global__ void __launch_bounds__(256)
gh_gemm(const unsigned short* __restrict__ xb, const unsigned short* __restrict__ la,
        const float* __restrict__ gates, unsigned short* __restrict__ ghout) {
  __shared__ __align__(16) unsigned short As[2][BM * LDT];
  __shared__ __align__(16) unsigned short Bs[2][BN * LDT];
  const int m0 = blockIdx.y * BM, n0 = blockIdx.x * BN;
  const int lane = threadIdx.x & 31, wid = threadIdx.x >> 5;
  const int wm = wid & 1, wn = wid >> 1;
  const int r = lane & 15, kh = lane >> 4;
  v8f acc[4][2];
#pragma unroll
  for (int mi = 0; mi < 4; ++mi)
#pragma unroll
    for (int ni = 0; ni < 2; ++ni) acc[mi][ni] = (v8f)(0.f);

  const unsigned short* gA = xb + (size_t)m0 * D_DIM;
  const unsigned short* gB = la + (size_t)n0 * D_DIM;
  if (wid == 0) {                       // prologue DMA into buffer 0
    tdm_tile(gA, D_DIM, (unsigned)(uintptr_t)&As[0][0]);
    tdm_tile(gB, D_DIM, (unsigned)(uintptr_t)&Bs[0][0]);
    TDM_WAIT();
  }
  __syncthreads();
  int buf = 0;
  for (int k0 = 0; k0 < D_DIM; k0 += BK, buf ^= 1) {
    if (wid == 0 && k0 + BK < D_DIM) {  // DMA next tile into other buffer (overlaps WMMA)
      tdm_tile(gA + k0 + BK, D_DIM, (unsigned)(uintptr_t)&As[buf ^ 1][0]);
      tdm_tile(gB + k0 + BK, D_DIM, (unsigned)(uintptr_t)&Bs[buf ^ 1][0]);
    }
    mma_step(As[buf], Bs[buf], wm, wn, r, kh, acc);
    if (wid == 0) TDM_WAIT();
    __syncthreads();
  }
#pragma unroll
  for (int mi = 0; mi < 4; ++mi)
#pragma unroll
    for (int ni = 0; ni < 2; ++ni) {
      int c = n0 + wn * 32 + ni * 16 + r;
      int e = c >> 5;  // 32 ranks per expert
#pragma unroll
      for (int j = 0; j < 8; ++j) {
        int t = m0 + wm * 64 + mi * 16 + 8 * kh + j;
        ghout[(size_t)t * C_DIM + c] = f2bf(gates[(size_t)t * E_EXP + e] * acc[mi][ni][j]);
      }
    }
}

// out = xb @ wb.T + bias + gh @ lbp.T   (K = 4096 then 256, same accumulators)
__global__ void __launch_bounds__(256)
moe_gemm(const unsigned short* __restrict__ xb, const unsigned short* __restrict__ wb,
         const unsigned short* __restrict__ gh, const unsigned short* __restrict__ lbp,
         const float* __restrict__ bias, float* __restrict__ out) {
  __shared__ __align__(16) unsigned short As[2][BM * LDT];
  __shared__ __align__(16) unsigned short Bs[2][BN * LDT];
  const int m0 = blockIdx.y * BM, n0 = blockIdx.x * BN;
  const int lane = threadIdx.x & 31, wid = threadIdx.x >> 5;
  const int wm = wid & 1, wn = wid >> 1;
  const int r = lane & 15, kh = lane >> 4;
  v8f acc[4][2];
#pragma unroll
  for (int mi = 0; mi < 4; ++mi)
#pragma unroll
    for (int ni = 0; ni < 2; ++ni) acc[mi][ni] = (v8f)(0.f);

  // ---- phase 1: base weights, K = 4096 ----
  {
    const unsigned short* gA = xb + (size_t)m0 * D_DIM;
    const unsigned short* gB = wb + (size_t)n0 * D_DIM;
    if (wid == 0) {
      tdm_tile(gA, D_DIM, (unsigned)(uintptr_t)&As[0][0]);
      tdm_tile(gB, D_DIM, (unsigned)(uintptr_t)&Bs[0][0]);
      TDM_WAIT();
    }
    __syncthreads();
    int buf = 0;
    for (int k0 = 0; k0 < D_DIM; k0 += BK, buf ^= 1) {
      if (wid == 0 && k0 + BK < D_DIM) {
        tdm_tile(gA + k0 + BK, D_DIM, (unsigned)(uintptr_t)&As[buf ^ 1][0]);
        tdm_tile(gB + k0 + BK, D_DIM, (unsigned)(uintptr_t)&Bs[buf ^ 1][0]);
      }
      mma_step(As[buf], Bs[buf], wm, wn, r, kh, acc);
      if (wid == 0) TDM_WAIT();
      __syncthreads();
    }
  }
  // ---- phase 2: LoRA, K = 256 on the same accumulators ----
  {
    const unsigned short* gA = gh  + (size_t)m0 * C_DIM;
    const unsigned short* gB = lbp + (size_t)n0 * C_DIM;
    if (wid == 0) {
      tdm_tile(gA, C_DIM, (unsigned)(uintptr_t)&As[0][0]);
      tdm_tile(gB, C_DIM, (unsigned)(uintptr_t)&Bs[0][0]);
      TDM_WAIT();
    }
    __syncthreads();
    int buf = 0;
    for (int k0 = 0; k0 < C_DIM; k0 += BK, buf ^= 1) {
      if (wid == 0 && k0 + BK < C_DIM) {
        tdm_tile(gA + k0 + BK, C_DIM, (unsigned)(uintptr_t)&As[buf ^ 1][0]);
        tdm_tile(gB + k0 + BK, C_DIM, (unsigned)(uintptr_t)&Bs[buf ^ 1][0]);
      }
      mma_step(As[buf], Bs[buf], wm, wn, r, kh, acc);
      if (wid == 0) TDM_WAIT();
      __syncthreads();
    }
  }
  // ---- epilogue: add bias, store f32 ----
#pragma unroll
  for (int ni = 0; ni < 2; ++ni) {
    int c = n0 + wn * 32 + ni * 16 + r;
    float bv = bias[c];
#pragma unroll
    for (int mi = 0; mi < 4; ++mi)
#pragma unroll
      for (int j = 0; j < 8; ++j) {
        int t = m0 + wm * 64 + mi * 16 + 8 * kh + j;
        out[(size_t)t * O_DIM + c] = acc[mi][ni][j] + bv;
      }
  }
}

// ---------------- host-side launch ----------------
extern "C" void kernel_launch(void* const* d_in, const int* in_sizes, int n_in,
                              void* d_out, int out_size, void* d_ws, size_t ws_size,
                              hipStream_t stream) {
  const float* x    = (const float*)d_in[0];
  const float* bw   = (const float*)d_in[1];
  const float* bb   = (const float*)d_in[2];
  const float* rw   = (const float*)d_in[3];
  const float* rb   = (const float*)d_in[4];
  const float* la_f = (const float*)d_in[5];
  const float* lb_f = (const float*)d_in[6];
  float* out = (float*)d_out;

  // workspace carve-up (~105 MB)
  char* p = (char*)d_ws;
  unsigned short* xb  = (unsigned short*)p; p += (size_t)T_TOK * D_DIM * 2;
  unsigned short* wbq = (unsigned short*)p; p += (size_t)O_DIM * D_DIM * 2;
  unsigned short* la  = (unsigned short*)p; p += (size_t)C_DIM * D_DIM * 2;
  unsigned short* lbp = (unsigned short*)p; p += (size_t)O_DIM * C_DIM * 2;
  unsigned short* gh  = (unsigned short*)p; p += (size_t)T_TOK * C_DIM * 2;
  float* gates        = (float*)p;          p += (size_t)T_TOK * E_EXP * 4;

  // 1) bf16 conversions / packing (one pass; thereafter GEMMs stream from L2)
  cvt_f32_bf16<<<8192, 256, 0, stream>>>(x, xb, (T_TOK * D_DIM) / 4);
  cvt_f32_bf16<<<4096, 256, 0, stream>>>(bw, wbq, (O_DIM * D_DIM) / 4);
  cvt_f32_bf16<<<1024, 256, 0, stream>>>(la_f, la, (C_DIM * D_DIM) / 4);
  pack_lora_b<<<(E_EXP * O_DIM * R_RANK + 255) / 256, 256, 0, stream>>>(lb_f, lbp);

  // 2) router gates (SCALING folded in)
  router_gates<<<T_TOK / 8, 256, 0, stream>>>(x, rw, rb, gates);

  // 3) rank activations with gating: gh[t, e*32+r]
  gh_gemm<<<dim3(C_DIM / BN, T_TOK / BM), 256, 0, stream>>>(xb, la, gates, gh);

  // 4) fused base + LoRA GEMM with bias
  moe_gemm<<<dim3(O_DIM / BN, T_TOK / BM), 256, 0, stream>>>(xb, wbq, gh, lbp, bb, out);
}
